// GIN_weighted_22625887715637
// MI455X (gfx1250) — matrix operand order repaired and checked by reading, not compiled, not run on hardware
//
#include <hip/hip_runtime.h>
#include <hip/hip_bf16.h>

typedef __attribute__((ext_vector_type(16))) __bf16 v16bf;
typedef __attribute__((ext_vector_type(8)))  float  v8f;

__device__ __forceinline__ v8f wmma_bf16(v16bf a, v16bf b, v8f c) {
  // D = A(16x32 bf16) * B(32x16 bf16) + C(16x16 f32)
  return __builtin_amdgcn_wmma_f32_16x16x32_bf16(
      /*neg_a=*/false, a, /*neg_b=*/false, b,
      /*c_mod=*/(short)0, c, /*reuse_a=*/false, /*reuse_b=*/false);
}

// A-fragment column for element i (contiguous K order, matching B fills, 16x32 tile):
//   lanes 0-15 (half=0): K = {0..15};  lanes 16-31 (half=1): K = {16..31}
__device__ __forceinline__ int a_col(int c, int i, int half) {
  return c * 32 + half * 16 + i;
}

// ---------------------------------------------------------------------------
// Edge MLP: per 16-edge tile:  m = relu((h[src]*ea) @ eW1 + eb1) @ eW2 + eb2
// then atomically scatter-add rows of m into aggr[dst].
// ---------------------------------------------------------------------------
__global__ void __launch_bounds__(256) edge_mlp_kernel(
    const float* __restrict__ h, const int* __restrict__ src,
    const int* __restrict__ dst, const float* __restrict__ ea,
    const float* __restrict__ eW1, const float* __restrict__ eb1,
    const float* __restrict__ eW2, const float* __restrict__ eb2,
    float* __restrict__ aggr, int E)
{
  __shared__ __bf16 smem[8][16][64];            // per-wave 16x64 bf16 staging
  const int lane  = threadIdx.x & 31;
  const int wq    = threadIdx.x >> 5;
  const int n     = lane & 15;
  const int half  = lane >> 4;
  const int wave  = (blockIdx.x * blockDim.x + threadIdx.x) >> 5;
  const int nwav  = (gridDim.x * blockDim.x) >> 5;
  const int ntile = (E + 15) >> 4;

  // Hoist weight B-fragments (loop-invariant): B[k][col], k = c*32 + half*16 + i
  v16bf B1[2][4], B2[2][4];
  float b1v[4], b2v[4];
  for (int c = 0; c < 2; ++c)
    for (int t = 0; t < 4; ++t)
      for (int i = 0; i < 16; ++i) {
        int k = c * 32 + half * 16 + i;
        B1[c][t][i] = (__bf16)eW1[k * 64 + t * 16 + n];
        B2[c][t][i] = (__bf16)eW2[k * 64 + t * 16 + n];
      }
  for (int t = 0; t < 4; ++t) { b1v[t] = eb1[t * 16 + n]; b2v[t] = eb2[t * 16 + n]; }

  for (int tile = wave; tile < ntile; tile += nwav) {
    int e  = tile * 16 + n;                      // A-row owned by this lane pair
    int eg = e < E ? e : E - 1;
    int s  = src[eg];
    float w = (e < E) ? ea[eg] : 0.0f;
    const float* hr = h + s * 64;

    v16bf A[2];
    for (int c = 0; c < 2; ++c)
      for (int i = 0; i < 16; ++i)
        A[c][i] = (__bf16)(hr[a_col(c, i, half)] * w);

    // GEMM1 + bias + ReLU -> LDS (D-layout -> memory)
    for (int t = 0; t < 4; ++t) {
      v8f C = {0.f,0.f,0.f,0.f,0.f,0.f,0.f,0.f};
      C = wmma_bf16(A[0], B1[0][t], C);
      C = wmma_bf16(A[1], B1[1][t], C);
      for (int r = 0; r < 8; ++r) {
        float v = C[r] + b1v[t];
        v = v > 0.f ? v : 0.f;
        smem[wq][r + 8 * half][t * 16 + n] = (__bf16)v;
      }
    }
    asm volatile("s_wait_dscnt 0x0" ::: "memory");

    // Reload in A-layout for GEMM2
    v16bf A2[2];
    for (int c = 0; c < 2; ++c)
      for (int i = 0; i < 16; ++i)
        A2[c][i] = smem[wq][n][a_col(c, i, half)];

    int didx[8];
    for (int r = 0; r < 8; ++r) {
      int er = tile * 16 + 8 * half + r;
      didx[r] = (er < E) ? dst[er] : -1;
    }

    // GEMM2 + bias + atomic scatter
    for (int t = 0; t < 4; ++t) {
      v8f C = {0.f,0.f,0.f,0.f,0.f,0.f,0.f,0.f};
      C = wmma_bf16(A2[0], B2[0][t], C);
      C = wmma_bf16(A2[1], B2[1][t], C);
      for (int r = 0; r < 8; ++r) {
        if (didx[r] >= 0) {
          float v = C[r] + b2v[t];
          unsafeAtomicAdd(&aggr[didx[r] * 64 + t * 16 + n], v);
        }
      }
    }
  }
}

// ---------------------------------------------------------------------------
// Node GEMM1:  t = (h + aggr) @ nW1 + nb1 ; also accumulate column sum/sumsq
// ---------------------------------------------------------------------------
__global__ void __launch_bounds__(256) node_gemm1_kernel(
    const float* __restrict__ hin, const float* __restrict__ aggr,
    const float* __restrict__ nW1, const float* __restrict__ nb1,
    float* __restrict__ tbuf, float* __restrict__ stats, int N)
{
  const int lane = threadIdx.x & 31;
  const int n    = lane & 15;
  const int half = lane >> 4;
  const int wave = (blockIdx.x * blockDim.x + threadIdx.x) >> 5;
  const int nwav = (gridDim.x * blockDim.x) >> 5;
  const int ntile = (N + 15) >> 4;

  v16bf B[2][4];
  float bias[4];
  for (int c = 0; c < 2; ++c)
    for (int t = 0; t < 4; ++t)
      for (int i = 0; i < 16; ++i) {
        int k = c * 32 + half * 16 + i;
        B[c][t][i] = (__bf16)nW1[k * 64 + t * 16 + n];
      }
  for (int t = 0; t < 4; ++t) bias[t] = nb1[t * 16 + n];

  float ssum[4] = {0,0,0,0}, ssq[4] = {0,0,0,0};

  for (int tile = wave; tile < ntile; tile += nwav) {
    int row = tile * 16 + n;
    int rg  = row < N ? row : N - 1;
    const float* hr = hin  + rg * 64;
    const float* ar = aggr + rg * 64;
    v16bf A[2];
    for (int c = 0; c < 2; ++c)
      for (int i = 0; i < 16; ++i) {
        int col = a_col(c, i, half);
        A[c][i] = (__bf16)(hr[col] + ar[col]);
      }
    for (int t = 0; t < 4; ++t) {
      v8f C = {0.f,0.f,0.f,0.f,0.f,0.f,0.f,0.f};
      C = wmma_bf16(A[0], B[0][t], C);
      C = wmma_bf16(A[1], B[1][t], C);
      for (int r = 0; r < 8; ++r) {
        int ro = tile * 16 + r + 8 * half;
        if (ro < N) {
          float v = C[r] + bias[t];
          tbuf[ro * 64 + t * 16 + n] = v;
          ssum[t] += v;
          ssq[t]  += v * v;
        }
      }
    }
  }
  for (int t = 0; t < 4; ++t) {
    unsafeAtomicAdd(&stats[t * 16 + n], ssum[t]);
    unsafeAtomicAdd(&stats[64 + t * 16 + n], ssq[t]);
  }
}

// Fold BN statistics into per-column scale/shift: y = x*scale + shift
__global__ void bn_finalize_kernel(const float* __restrict__ stats,
                                   const float* __restrict__ g,
                                   const float* __restrict__ b,
                                   float* __restrict__ bn, float invn)
{
  int c = threadIdx.x;            // 64 threads
  float mu  = stats[c] * invn;
  float var = stats[64 + c] * invn - mu * mu;
  float sc  = g[c] * rsqrtf(var + 1e-5f);
  bn[c]      = sc;
  bn[64 + c] = b[c] - mu * sc;
}

// ---------------------------------------------------------------------------
// Node GEMM2:  h = relu( relu(bn(t)) @ nW2 + nb2 )
// ---------------------------------------------------------------------------
__global__ void __launch_bounds__(256) node_gemm2_kernel(
    const float* __restrict__ tbuf, const float* __restrict__ bn,
    const float* __restrict__ nW2, const float* __restrict__ nb2,
    float* __restrict__ hout, int N)
{
  const int lane = threadIdx.x & 31;
  const int n    = lane & 15;
  const int half = lane >> 4;
  const int wave = (blockIdx.x * blockDim.x + threadIdx.x) >> 5;
  const int nwav = (gridDim.x * blockDim.x) >> 5;
  const int ntile = (N + 15) >> 4;

  v16bf B[2][4];
  float bias[4];
  for (int c = 0; c < 2; ++c)
    for (int t = 0; t < 4; ++t)
      for (int i = 0; i < 16; ++i) {
        int k = c * 32 + half * 16 + i;
        B[c][t][i] = (__bf16)nW2[k * 64 + t * 16 + n];
      }
  for (int t = 0; t < 4; ++t) bias[t] = nb2[t * 16 + n];

  for (int tile = wave; tile < ntile; tile += nwav) {
    int row = tile * 16 + n;
    int rg  = row < N ? row : N - 1;
    const float* tr = tbuf + rg * 64;
    v16bf A[2];
    for (int c = 0; c < 2; ++c)
      for (int i = 0; i < 16; ++i) {
        int col = a_col(c, i, half);
        float v = tr[col] * bn[col] + bn[64 + col];
        A[c][i] = (__bf16)(v > 0.f ? v : 0.f);
      }
    for (int t = 0; t < 4; ++t) {
      v8f C = {0.f,0.f,0.f,0.f,0.f,0.f,0.f,0.f};
      C = wmma_bf16(A[0], B[0][t], C);
      C = wmma_bf16(A[1], B[1][t], C);
      for (int r = 0; r < 8; ++r) {
        int ro = tile * 16 + r + 8 * half;
        if (ro < N) {
          float v = C[r] + bias[t];
          hout[ro * 64 + t * 16 + n] = v > 0.f ? v : 0.f;
        }
      }
    }
  }
}

// Mean-pool: atomic accumulate per-graph sums and counts
__global__ void pool_kernel(const float* __restrict__ h, const int* __restrict__ batch,
                            float* __restrict__ pool, int* __restrict__ cnt, int N)
{
  int idx = blockIdx.x * blockDim.x + threadIdx.x;
  if (idx < N * 64) {
    int node = idx >> 6, col = idx & 63;
    int bg = batch[node];
    unsafeAtomicAdd(&pool[bg * 64 + col], h[idx]);
    if (col == 0) atomicAdd(&cnt[bg], 1);
  }
}

// Final head: out = relu(bn(pooled @ W1 + b1)) @ W2 + b2 ; one block of G threads
__global__ void final_head_kernel(
    const float* __restrict__ pool, const int* __restrict__ cnt,
    const float* __restrict__ W1, const float* __restrict__ b1,
    const float* __restrict__ g, const float* __restrict__ b,
    const float* __restrict__ W2, const float* __restrict__ b2,
    float* __restrict__ out, int G)
{
  __shared__ float s_sum[64], s_sq[64], s_scale[64], s_shift[64];
  int tid = threadIdx.x;                 // one thread per graph
  if (tid < 64) { s_sum[tid] = 0.f; s_sq[tid] = 0.f; }
  __syncthreads();

  float p[64], t2[64];
  float c = (float)cnt[tid]; c = c > 1.f ? c : 1.f;
  for (int i = 0; i < 64; ++i) p[i] = pool[tid * 64 + i] / c;
  for (int o = 0; o < 64; ++o) {
    float acc = b1[o];
    for (int k = 0; k < 64; ++k) acc += p[k] * W1[k * 64 + o];
    t2[o] = acc;
    atomicAdd(&s_sum[o], acc);
    atomicAdd(&s_sq[o], acc * acc);
  }
  __syncthreads();
  if (tid < 64) {
    float mu  = s_sum[tid] / (float)G;
    float var = s_sq[tid] / (float)G - mu * mu;
    float sc  = g[tid] * rsqrtf(var + 1e-5f);
    s_scale[tid] = sc;
    s_shift[tid] = b[tid] - mu * sc;
  }
  __syncthreads();
  float y[64];
  for (int o = 0; o < 64; ++o) {
    float v = t2[o] * s_scale[o] + s_shift[o];
    y[o] = v > 0.f ? v : 0.f;
  }
  for (int o = 0; o < 32; ++o) {
    float acc = b2[o];
    for (int k = 0; k < 64; ++k) acc += y[k] * W2[k * 32 + o];
    out[tid * 32 + o] = acc;
  }
}

extern "C" void kernel_launch(void* const* d_in, const int* in_sizes, int n_in,
                              void* d_out, int out_size, void* d_ws, size_t ws_size,
                              hipStream_t stream)
{
  const float* x   = (const float*)d_in[0];
  const int* eidx  = (const int*)d_in[1];
  const float* ea  = (const float*)d_in[2];
  const int* batch = (const int*)d_in[3];
  const int E = in_sizes[2];
  const int N = in_sizes[3];
  const int G = out_size / 32;
  const int* srcp = eidx;
  const int* dstp = eidx + E;

  float* ws_h     = (float*)d_ws;
  float* ws_aggr  = ws_h + (size_t)N * 64;
  float* ws_t     = ws_aggr + (size_t)N * 64;
  float* ws_stats = ws_t + (size_t)N * 64;
  float* ws_bn    = ws_stats + 128;
  float* ws_pool  = ws_bn + 128;
  int*   ws_cnt   = (int*)(ws_pool + (size_t)G * 64);

  const float* hin = x;
  for (int l = 0; l < 2; ++l) {
    void* const* wp = d_in + 4 + 10 * l;
    const float *eW1 = (const float*)wp[0], *eb1 = (const float*)wp[1];
    const float *eW2 = (const float*)wp[2], *eb2 = (const float*)wp[3];
    const float *nW1 = (const float*)wp[4], *nb1 = (const float*)wp[5];
    const float *gg  = (const float*)wp[6], *bb  = (const float*)wp[7];
    const float *nW2 = (const float*)wp[8], *nb2 = (const float*)wp[9];

    hipMemsetAsync(ws_aggr, 0, (size_t)N * 64 * sizeof(float), stream);
    edge_mlp_kernel<<<1024, 256, 0, stream>>>(hin, srcp, dstp, ea,
                                              eW1, eb1, eW2, eb2, ws_aggr, E);
    hipMemsetAsync(ws_stats, 0, 128 * sizeof(float), stream);
    node_gemm1_kernel<<<512, 256, 0, stream>>>(hin, ws_aggr, nW1, nb1,
                                               ws_t, ws_stats, N);
    bn_finalize_kernel<<<1, 64, 0, stream>>>(ws_stats, gg, bb, ws_bn, 1.0f / (float)N);
    node_gemm2_kernel<<<512, 256, 0, stream>>>(ws_t, ws_bn, nW2, nb2, ws_h, N);
    hin = ws_h;
  }

  hipMemsetAsync(ws_pool, 0, (size_t)G * 64 * sizeof(float) + G * sizeof(int), stream);
  pool_kernel<<<((N * 64) + 255) / 256, 256, 0, stream>>>(ws_h, batch, ws_pool, ws_cnt, N);

  const float *fW1 = (const float*)d_in[24], *fb1 = (const float*)d_in[25];
  const float *fg  = (const float*)d_in[26], *fb  = (const float*)d_in[27];
  const float *fW2 = (const float*)d_in[28], *fb2 = (const float*)d_in[29];
  final_head_kernel<<<1, G, 0, stream>>>(ws_pool, ws_cnt, fW1, fb1, fg, fb,
                                         fW2, fb2, (float*)d_out, G);
}
